// MoEGenomeLayer_80994493268350
// MI455X (gfx1250) — matrix operand order repaired and checked by reading, not compiled, not run on hardware
//
#include <hip/hip_runtime.h>
#include <hip/hip_bf16.h>

// ---------------------------------------------------------------------------
// MoE layer, MI455X (gfx1250, wave32, WMMA bf16 16x16x32, f32 accumulate)
// B=8 T=4096 D=1024 E=8 H=64 K=2  -> N=32768 tokens
// Sparse (top-2) expert compute ~25.8 GFLOP; mandatory HBM traffic ~256 MB
// => memory-roofline bound (~11us @ 23.3 TB/s) if matmuls run on bf16 WMMA.
// Router fuses x->bf16 (xb, 64MB, L2-resident) so phase A gathers bf16
// instead of fp32; all bf16 tile staging uses GLOBAL_LOAD_ASYNC_TO_LDS_B128
// (ASYNCcnt) when the builtin is available.
// ---------------------------------------------------------------------------

typedef __attribute__((ext_vector_type(16))) __bf16 bfvec16;
typedef __attribute__((ext_vector_type(8)))  float  fvec8;
typedef __attribute__((ext_vector_type(4)))  int    iv4;   // builtin's pointee type

#define NTOK   32768
#define DDIM   1024
#define NEXP   8
#define HDIM   64

#if __has_builtin(__builtin_amdgcn_global_load_async_to_lds_b128)
#define HAVE_ASYNC_LDS 1
#endif

// 16-byte global -> LDS copy; async (ASYNCcnt) when the builtin exists.
// Probe-learned signature: (v4i* global_src, v4i* lds_dst, imm offset, imm cpol)
// with generic pointers (clang inserts the address-space casts).
__device__ __forceinline__ void cp16_g2l(const void* g, void* l) {
#ifdef HAVE_ASYNC_LDS
    __builtin_amdgcn_global_load_async_to_lds_b128((iv4*)g, (iv4*)l, 0, 0);
#else
    *reinterpret_cast<uint4*>(l) = *reinterpret_cast<const uint4*>(g);
#endif
}

__device__ __forceinline__ void async_drain() {
#ifdef HAVE_ASYNC_LDS
#if __has_builtin(__builtin_amdgcn_s_wait_asynccnt)
    __builtin_amdgcn_s_wait_asynccnt(0);
#else
    asm volatile("s_wait_asynccnt 0x0" ::: "memory");
#endif
#endif
}

// round-to-nearest-even float -> bf16 bits
__device__ __forceinline__ unsigned short f2bf(float f) {
    union { float f; unsigned u; } c; c.f = f;
    unsigned u = c.u;
    unsigned r = u + 0x7FFFu + ((u >> 16) & 1u);
    return (unsigned short)(r >> 16);
}

// A fragment: 16x32 bf16 tile, LDS row-major, strideU32 = row stride in u32.
// Per ISA 7.12.2: lane<16 -> M=lane, K pairs {0..7},{16..23};
//                 lane>=16 -> M=lane-16, K pairs {8..15},{24..31}.
__device__ __forceinline__ bfvec16 afrag(const unsigned* lds, int m, int strideU32, int half) {
    union { bfvec16 v; unsigned u[8]; } f;
    const unsigned* row = lds + m * strideU32;
    int o = 4 * half;
#pragma unroll
    for (int j = 0; j < 4; ++j) { f.u[j] = row[o + j]; f.u[4 + j] = row[8 + o + j]; }
    return f.v;
}

// B fragment: 32x16 bf16 tile stored column-major in LDS ([n][k]).
// lanes 0-15: N=lane, K {0..7},{16..23} ; lanes 16-31: N=lane-16, K {8..15},{24..31}.
__device__ __forceinline__ bfvec16 bfrag(const unsigned* lds, int n, int strideU32, int half) {
    union { bfvec16 v; unsigned u[8]; } f;
    const unsigned* row = lds + n * strideU32 + 4 * half;
#pragma unroll
    for (int j = 0; j < 4; ++j) { f.u[j] = row[j]; f.u[4 + j] = row[8 + j]; }
    return f.v;
}

// ---------------------------------------------------------------------------
// 0) zero output + expert counters (deterministic each call)
// ---------------------------------------------------------------------------
__global__ void __launch_bounds__(256) k_zero(float* out, int n, int* cnt) {
    int i = blockIdx.x * 256 + threadIdx.x;
    if (i < NEXP) cnt[i] = 0;
    int stride = gridDim.x * 256;
    for (int k = i; k < n; k += stride) out[k] = 0.f;
}

// ---------------------------------------------------------------------------
// 1) weight prep: bf16 + transpose for WMMA-friendly B tiles
//    downT/gateT : [E][H][D], upT : [E][D][H]
// ---------------------------------------------------------------------------
__global__ void __launch_bounds__(256) k_prep(const float* ed, const float* eg, const float* eu,
                                              unsigned short* dT, unsigned short* gT,
                                              unsigned short* uT) {
    int i = blockIdx.x * 256 + threadIdx.x;          // over E*D*H = 524288
    if (i >= NEXP * DDIM * HDIM) return;
    int e   = i >> 16;                               // D*H = 65536 per expert
    int rem = i & 65535;
    int h = rem >> 10, d = rem & 1023;               // i == (e*64+h)*1024 + d
    dT[i] = f2bf(ed[((size_t)e * DDIM + d) * HDIM + h]);
    gT[i] = f2bf(eg[((size_t)e * DDIM + d) * HDIM + h]);
    int d2 = rem >> 6, h2 = rem & 63;                // i == (e*1024+d2)*64 + h2
    uT[i] = f2bf(eu[((size_t)e * HDIM + h2) * DDIM + d2]);
}

// ---------------------------------------------------------------------------
// 2) router: one wave per token. 8 dots of len 1024, top-2 + softmax,
//    bucket slot via atomicAdd (slot value = token*2+k, scale folded in w).
//    Optionally also emits xb = bf16(x) so phase A gathers bf16.
// ---------------------------------------------------------------------------
__global__ void __launch_bounds__(256) k_router(const float* x, const float* rw,
                                                const float* scale_p,
                                                int* cnt, int* slots, float* slotw,
                                                unsigned short* xb, int do_xb) {
    int wave = threadIdx.x >> 5, lane = threadIdx.x & 31;
    int token = blockIdx.x * 8 + wave;
    const float* xr = x + (size_t)token * DDIM;
    float acc[NEXP];
#pragma unroll
    for (int e = 0; e < NEXP; ++e) acc[e] = 0.f;
    for (int c = lane; c < DDIM; c += 32) {
        float xv = xr[c];
        if (do_xb) xb[(size_t)token * DDIM + c] = f2bf(xv);
#pragma unroll
        for (int e = 0; e < NEXP; ++e) acc[e] += xv * rw[e * DDIM + c];
    }
#pragma unroll
    for (int e = 0; e < NEXP; ++e)
        for (int off = 16; off > 0; off >>= 1) acc[e] += __shfl_xor(acc[e], off, 32);
    if (lane == 0) {
        int i0 = 0; float v0 = acc[0];
        for (int e = 1; e < NEXP; ++e) if (acc[e] > v0) { v0 = acc[e]; i0 = e; }
        int i1 = (i0 == 0) ? 1 : 0; float v1 = acc[i1];
        for (int e = 0; e < NEXP; ++e) if (e != i0 && acc[e] > v1) { v1 = acc[e]; i1 = e; }
        float w1 = __expf(v1 - v0);
        float inv = 1.f / (1.f + w1);
        float w0 = inv; w1 *= inv;
        float sc = scale_p[0];
        int s0 = atomicAdd(&cnt[i0], 1);
        slots[i0 * NTOK + s0] = token * 2 + 0;
        slotw[i0 * NTOK + s0] = w0 * sc;
        int s1 = atomicAdd(&cnt[i1], 1);
        slots[i1 * NTOK + s1] = token * 2 + 1;
        slotw[i1 * NTOK + s1] = w1 * sc;
    }
}

// ---------------------------------------------------------------------------
// 3) phase A: per-expert GEMM [cnt,1024] x [1024,64] for down & gate.
//    128-token tile / block; 8 waves x (16-row M-subtile x 4 N-tiles).
//    BF16X=1: gather bf16 rows from xb via async-to-LDS.
//    BF16X=0: gather fp32 rows from x and convert while staging.
//    Epilogue: hidden[slot] = bf16( silu(g) * u * (cw*scale) )
// ---------------------------------------------------------------------------
template <bool BF16X>
__global__ void __launch_bounds__(256) k_phaseA(const float* x, const unsigned short* xb,
                                                const unsigned short* downT,
                                                const unsigned short* gateT,
                                                const int* cnt, const int* slots,
                                                const float* slotw,
                                                unsigned short* hidden) {
    const int e    = blockIdx.x >> 8;     // 256 tiles per expert (worst case)
    const int tile = blockIdx.x & 255;
    const int c    = cnt[e];
    const int row0 = tile * 128;
    if (row0 >= c) return;

    __shared__ unsigned short As[128 * 40];   // 128 x 32 bf16, padded
    __shared__ unsigned short Bd[64 * 40];    // 64(n) x 32(k) bf16, padded
    __shared__ unsigned short Bg[64 * 40];
    __shared__ int   Srow[128];
    __shared__ float Wrow[128];

    const int t = threadIdx.x;
    if (t < 128) {
        int gi = row0 + t;
        int ok = gi < c;
        Srow[t] = ok ? slots[e * NTOK + gi] : -1;
        Wrow[t] = ok ? slotw[e * NTOK + gi] : 0.f;
    }
    __syncthreads();

    const int wave = t >> 5, lane = t & 31, half = lane >> 4, l15 = lane & 15;
    fvec8 accd[4], accu[4];
#pragma unroll
    for (int nt = 0; nt < 4; ++nt) { accd[nt] = (fvec8)0.f; accu[nt] = (fvec8)0.f; }

    for (int k0 = 0; k0 < DDIM; k0 += 32) {
        __syncthreads();
        // ---- stage A tile (128 x 32 bf16) ----
        if (BF16X) {
            // 512 16B-chunks: idx -> (row, 16B chunk), async gather from xb
#pragma unroll
            for (int pass = 0; pass < 2; ++pass) {
                int idx = t + pass * 256;
                int rr = idx >> 2, q = idx & 3;
                int s = Srow[rr];
                if (s >= 0)
                    cp16_g2l(xb + (size_t)(s >> 1) * DDIM + k0 + 8 * q, &As[rr * 40 + 8 * q]);
                else
                    *reinterpret_cast<uint4*>(&As[rr * 40 + 8 * q]) = make_uint4(0u, 0u, 0u, 0u);
            }
        } else {
            // fp32 path: 8 threads/row, 4 floats each, convert to bf16
            int r = t >> 3, colq = (t & 7) * 4;
            for (int rr = r; rr < 128; rr += 32) {
                int s = Srow[rr];
                float4 v = make_float4(0.f, 0.f, 0.f, 0.f);
                if (s >= 0)
                    v = *reinterpret_cast<const float4*>(x + (size_t)(s >> 1) * DDIM + k0 + colq);
                unsigned lo = ((unsigned)f2bf(v.y) << 16) | f2bf(v.x);
                unsigned hi = ((unsigned)f2bf(v.w) << 16) | f2bf(v.z);
                unsigned* dst = reinterpret_cast<unsigned*>(&As[rr * 40 + colq]);
                dst[0] = lo; dst[1] = hi;
            }
        }
        // ---- stage Bd/Bg tiles (64 x 32 bf16 each), async ----
        {
            int idx = t * 4, n = idx >> 4, j = idx & 15;   // j in {0,4,8,12}
            size_t gb = ((size_t)(e * HDIM + n)) * DDIM + k0 + 2 * j;
            cp16_g2l(downT + gb, &Bd[n * 40 + 2 * j]);
            cp16_g2l(gateT + gb, &Bg[n * 40 + 2 * j]);
        }
        async_drain();
        __syncthreads();

        bfvec16 a = afrag(reinterpret_cast<const unsigned*>(As), wave * 16 + l15, 20, half);
#pragma unroll
        for (int nt = 0; nt < 4; ++nt) {
            bfvec16 bd = bfrag(reinterpret_cast<const unsigned*>(Bd), nt * 16 + l15, 20, half);
            accd[nt] = __builtin_amdgcn_wmma_f32_16x16x32_bf16(false, a, false, bd,
                                                               (short)0, accd[nt], false, false);
            bfvec16 bg = bfrag(reinterpret_cast<const unsigned*>(Bg), nt * 16 + l15, 20, half);
            accu[nt] = __builtin_amdgcn_wmma_f32_16x16x32_bf16(false, a, false, bg,
                                                               (short)0, accu[nt], false, false);
        }
    }

    // epilogue: silu(g)*u*w -> bf16 hidden[slot][col]
#pragma unroll
    for (int nt = 0; nt < 4; ++nt) {
        int col = nt * 16 + l15;
#pragma unroll
        for (int r = 0; r < 8; ++r) {
            int mloc = wave * 16 + r + 8 * half;
            int s = Srow[mloc];
            if (s >= 0) {
                float g = accd[nt][r];
                float u = accu[nt][r];
                float h = (g / (1.f + __expf(-g))) * u * Wrow[mloc];
                hidden[(size_t)s * HDIM + col] = f2bf(h);
            }
        }
    }
}

// ---------------------------------------------------------------------------
// 4) phase B: per-expert GEMM [cnt,64] x [64,1024]; scatter-accumulate into
//    out with f32 atomics (each token gets exactly 2 contributions).
//    grid: (16 d-tiles of 64, 256 token-tiles, 8 experts)
// ---------------------------------------------------------------------------
__global__ void __launch_bounds__(256) k_phaseB(const unsigned short* hidden,
                                                const unsigned short* upT,
                                                const int* cnt, const int* slots,
                                                float* out) {
    const int e    = blockIdx.z;
    const int tile = blockIdx.y;
    const int d0   = blockIdx.x * 64;
    const int c    = cnt[e];
    const int row0 = tile * 128;
    if (row0 >= c) return;

    __shared__ unsigned short As[128 * 40];
    __shared__ unsigned short Bs[64 * 40];
    __shared__ int Srow[128];

    const int t = threadIdx.x;
    if (t < 128) {
        int gi = row0 + t;
        Srow[t] = (gi < c) ? slots[e * NTOK + gi] : -1;
    }
    __syncthreads();

    const int wave = t >> 5, lane = t & 31, half = lane >> 4, l15 = lane & 15;
    fvec8 acc[4];
#pragma unroll
    for (int nt = 0; nt < 4; ++nt) acc[nt] = (fvec8)0.f;

    for (int k0 = 0; k0 < HDIM; k0 += 32) {
        __syncthreads();
        // stage A (bf16 hidden rows): 512 16B-chunks, async gather
#pragma unroll
        for (int pass = 0; pass < 2; ++pass) {
            int idx = t + pass * 256;
            int rr = idx >> 2, q = idx & 3;
            int s = Srow[rr];
            if (s >= 0)
                cp16_g2l(hidden + (size_t)s * HDIM + k0 + 8 * q, &As[rr * 40 + 8 * q]);
            else
                *reinterpret_cast<uint4*>(&As[rr * 40 + 8 * q]) = make_uint4(0u, 0u, 0u, 0u);
        }
        // stage B from upT[e][d0+n][k0..k0+31], async
        {
            int idx = t * 4, n = idx >> 4, j = idx & 15;
            cp16_g2l(upT + ((size_t)e * DDIM + d0 + n) * HDIM + k0 + 2 * j,
                     &Bs[n * 40 + 2 * j]);
        }
        async_drain();
        __syncthreads();

        bfvec16 a = afrag(reinterpret_cast<const unsigned*>(As), wave * 16 + l15, 20, half);
#pragma unroll
        for (int nt = 0; nt < 4; ++nt) {
            bfvec16 b = bfrag(reinterpret_cast<const unsigned*>(Bs), nt * 16 + l15, 20, half);
            acc[nt] = __builtin_amdgcn_wmma_f32_16x16x32_bf16(false, a, false, b,
                                                              (short)0, acc[nt], false, false);
        }
    }

#pragma unroll
    for (int nt = 0; nt < 4; ++nt) {
        int d = d0 + nt * 16 + l15;
#pragma unroll
        for (int r = 0; r < 8; ++r) {
            int mloc = wave * 16 + r + 8 * half;
            int s = Srow[mloc];
            if (s >= 0)
                atomicAdd(&out[(size_t)(s >> 1) * DDIM + d], acc[nt][r]);
        }
    }
}

// ---------------------------------------------------------------------------
// launch
// ---------------------------------------------------------------------------
extern "C" void kernel_launch(void* const* d_in, const int* in_sizes, int n_in,
                              void* d_out, int out_size, void* d_ws, size_t ws_size,
                              hipStream_t stream) {
    const float* x     = (const float*)d_in[0];
    const float* rw    = (const float*)d_in[1];
    const float* ed    = (const float*)d_in[2];   // expert_down  [E,D,H]
    const float* eu    = (const float*)d_in[3];   // expert_up    [E,H,D]
    const float* eg    = (const float*)d_in[4];   // expert_gate  [E,D,H]
    const float* scale = (const float*)d_in[5];
    float* out = (float*)d_out;

    // workspace layout
    char* w = (char*)d_ws;
    size_t o = 0;
    int* cnt = (int*)(w + o);                          o += 256;
    int* slots = (int*)(w + o);                        o += (size_t)NEXP * NTOK * 4;       // 1 MB
    float* slotw = (float*)(w + o);                    o += (size_t)NEXP * NTOK * 4;       // 1 MB
    unsigned short* hidden = (unsigned short*)(w + o); o += (size_t)NTOK * 2 * HDIM * 2;   // 8 MB
    unsigned short* dT = (unsigned short*)(w + o);     o += (size_t)NEXP * HDIM * DDIM * 2; // 1 MB
    unsigned short* gT = (unsigned short*)(w + o);     o += (size_t)NEXP * HDIM * DDIM * 2; // 1 MB
    unsigned short* uT = (unsigned short*)(w + o);     o += (size_t)NEXP * DDIM * HDIM * 2; // 1 MB
    unsigned short* xb = (unsigned short*)(w + o);
    size_t need_xb = o + (size_t)NTOK * DDIM * 2;      // +64 MB
    const bool use_xb = ws_size >= need_xb;            // deterministic host-side choice
    (void)in_sizes; (void)n_in; (void)out_size;

    k_zero<<<4096, 256, 0, stream>>>(out, NTOK * DDIM, cnt);
    k_prep<<<(NEXP * DDIM * HDIM + 255) / 256, 256, 0, stream>>>(ed, eg, eu, dT, gT, uT);
    k_router<<<NTOK / 8, 256, 0, stream>>>(x, rw, scale, cnt, slots, slotw, xb, use_xb ? 1 : 0);
    if (use_xb)
        k_phaseA<true><<<NEXP * 256, 256, 0, stream>>>(x, xb, dT, gT, cnt, slots, slotw, hidden);
    else
        k_phaseA<false><<<NEXP * 256, 256, 0, stream>>>(x, xb, dT, gT, cnt, slots, slotw, hidden);
    k_phaseB<<<dim3(DDIM / 64, 256, NEXP), 256, 0, stream>>>(hidden, uT, cnt, slots, out);
}
